// SAT_43267500540233
// MI455X (gfx1250) — matrix operand order repaired and checked by reading, compiled, not run on hardware
//
#include <hip/hip_runtime.h>
#include <hip/hip_bf16.h>
#include <math.h>
#include <stdint.h>

typedef __attribute__((ext_vector_type(16))) __bf16 v16bf;
typedef __attribute__((ext_vector_type(8)))  float  v8f;
typedef unsigned int u32x4 __attribute__((ext_vector_type(4)));
typedef int          i32x8 __attribute__((ext_vector_type(8)));
typedef int          i32x4 __attribute__((ext_vector_type(4)));

#define D_    768
#define NH_   12
#define HD_   64
#define B_    32
#define CIN_  512
#define HW_   16
#define L_    341
#define LP_   352
#define OUTD_ 512

#if defined(__has_builtin)
# if __has_builtin(__builtin_amdgcn_tensor_load_to_lds) && __has_builtin(__builtin_amdgcn_s_wait_tensorcnt)
#  define USE_TDM 1
# endif
#endif
#ifndef USE_TDM
# define USE_TDM 0
#endif

// ---------- helpers ----------
__device__ __forceinline__ float bf2f(unsigned short h){
    unsigned int u = ((unsigned int)h) << 16;
    float f; __builtin_memcpy(&f, &u, 4); return f;
}
__device__ __forceinline__ unsigned short f2bf(float f){
    unsigned int u; __builtin_memcpy(&u, &f, 4);
    u += 0x7fffu + ((u >> 16) & 1u);
    return (unsigned short)(u >> 16);
}

union FragU  { uint4 u[2]; v16bf v; };
union Frag16 { unsigned short s[16]; v16bf v; };

__device__ __forceinline__ v16bf make_frag(uint4 a, uint4 b){
    FragU f; f.u[0] = a; f.u[1] = b; return f.v;
}
__device__ __forceinline__ v8f wmma_bf16(v16bf a, v16bf b, v8f c){
    return __builtin_amdgcn_wmma_f32_16x16x32_bf16(false, a, false, b, (short)0, c, false, false);
}
__device__ __forceinline__ int lvl_end(int q){
    if (q < 1)  return 1;
    if (q < 5)  return 5;
    if (q < 21) return 21;
    if (q < 85) return 85;
    return 341;
}

// ---------- f32 -> bf16 convert ----------
__global__ void f2bf_kernel(const float* __restrict__ in, unsigned short* __restrict__ out, int n){
    int i = blockIdx.x * 256 + threadIdx.x;
    if (i < n) out[i] = f2bf(in[i]);
}

// ---------- style spatial mean ----------
__global__ void style_pool_kernel(const float* __restrict__ sf, unsigned short* __restrict__ out){
    int i = blockIdx.x * 256 + threadIdx.x;
    if (i >= B_ * CIN_) return;
    const float* p = sf + (size_t)i * HW_ * HW_;
    float s = 0.f;
    for (int j = 0; j < HW_ * HW_; ++j) s += p[j];
    out[i] = f2bf(s * (1.0f / (HW_ * HW_)));
}

// ---------- silu ----------
__global__ void silu_kernel(const float* __restrict__ in, unsigned short* __restrict__ out, int n){
    int i = blockIdx.x * 256 + threadIdx.x;
    if (i < n){ float x = in[i]; out[i] = f2bf(x / (1.0f + __expf(-x))); }
}

// ---------- patch avg-pool (writes token-major bf16) ----------
__global__ void pool_kernel(const float* __restrict__ content, unsigned short* __restrict__ out,
                            int pn, int rowbase){
    int r = HW_ / pn;
    int T = pn * pn;
    int total = B_ * T * CIN_;
    int i = blockIdx.x * 256 + threadIdx.x;
    if (i >= total) return;
    int c = i % CIN_;
    int t = (i / CIN_) % T;
    int b = i / (CIN_ * T);
    int py = t / pn, px = t % pn;
    const float* p = content + (((size_t)b * CIN_ + c) * HW_ + py * r) * HW_ + px * r;
    float s = 0.f;
    for (int dy = 0; dy < r; ++dy)
        for (int dx = 0; dx < r; ++dx)
            s += p[dy * HW_ + dx];
    out[((size_t)(rowbase + b * T + t)) * CIN_ + c] = f2bf(s / (float)(r * r));
}

// ---------- LayerNorm (+ optional adaLN modulation), bf16 out ----------
__global__ void ln_kernel(const float* __restrict__ x, const float* __restrict__ w,
                          const float* __restrict__ bb, const float* __restrict__ mod,
                          int soff, int shoff, unsigned short* __restrict__ out,
                          int tokens_per_b, int tok_off, float eps){
    __shared__ float r1[256], r2[256];
    int rid = blockIdx.x;
    int b = rid / tokens_per_b, t = rid - b * tokens_per_b;
    const float* xr = x + ((size_t)b * L_ + tok_off + t) * D_;
    float s1 = 0.f, s2 = 0.f;
    for (int c = threadIdx.x; c < D_; c += 256){ float v = xr[c]; s1 += v; s2 += v * v; }
    r1[threadIdx.x] = s1; r2[threadIdx.x] = s2; __syncthreads();
    for (int s = 128; s > 0; s >>= 1){
        if (threadIdx.x < s){ r1[threadIdx.x] += r1[threadIdx.x + s]; r2[threadIdx.x] += r2[threadIdx.x + s]; }
        __syncthreads();
    }
    float mu  = r1[0] * (1.0f / D_);
    float var = r2[0] * (1.0f / D_) - mu * mu;
    float inv = rsqrtf(var + eps);
    unsigned short* orow = out + (size_t)rid * D_;
    for (int c = threadIdx.x; c < D_; c += 256){
        float v = (xr[c] - mu) * inv * w[c] + bb[c];
        if (mod){
            float sc = mod[(size_t)b * 3072 + soff + c];
            float sh = mod[(size_t)b * 3072 + shoff + c];
            v = v * (1.0f + sc) + sh;
        }
        orow[c] = f2bf(v);
    }
}

// ---------- RoPE + (B,L,3,NH,HD) -> (B,NH,L,HD) transpose, bf16 ----------
__global__ void rope_kernel(const unsigned short* __restrict__ qkv,
                            unsigned short* __restrict__ qo,
                            unsigned short* __restrict__ ko,
                            unsigned short* __restrict__ vo){
    int token = blockIdx.x;
    int b = token / L_, l = token - b * L_;
    int p = threadIdx.x;                 // 0..383
    int head = p >> 5, pi = p & 31, d0 = pi * 2;
    const unsigned short* base = qkv + (size_t)token * (3 * D_);
    float fr  = __expf(-(float)d0 * (9.2103403719761836f / 64.0f)); // 10000^(-d0/64)
    float ang = (float)l * fr;
    float sn = __sinf(ang), cs = __cosf(ang);
    size_t o = (((size_t)b * NH_ + head) * L_ + l) * HD_ + d0;
    float x0 = bf2f(base[head * HD_ + d0]);
    float x1 = bf2f(base[head * HD_ + d0 + 1]);
    qo[o]     = f2bf(x0 * cs - x1 * sn);
    qo[o + 1] = f2bf(x1 * cs + x0 * sn);
    x0 = bf2f(base[D_ + head * HD_ + d0]);
    x1 = bf2f(base[D_ + head * HD_ + d0 + 1]);
    ko[o]     = f2bf(x0 * cs - x1 * sn);
    ko[o + 1] = f2bf(x1 * cs + x0 * sn);
    vo[o]     = base[2 * D_ + head * HD_ + d0];
    vo[o + 1] = base[2 * D_ + head * HD_ + d0 + 1];
}

// ---------- generic WMMA GEMM: out = A(MxK) . W(NxK)^T + bias ----------
// epi: 0 = none, 1 = exact gelu. Optional bias2 (per-col), residual (fp32, same idx).
// If tokens_per_b > 0, output row m maps to (m/tpb)*L_ + tok_off + (m%tpb).
__global__ void gemm_bf16_kernel(const unsigned short* __restrict__ A, int lda,
                                 const unsigned short* __restrict__ W, int ldw,
                                 float* __restrict__ Cf, unsigned short* __restrict__ Cb, int ldc,
                                 const float* __restrict__ bias, const float* __restrict__ bias2,
                                 const float* __restrict__ res,
                                 int M, int N, int K, int epi, int tokens_per_b, int tok_off){
    int tid  = threadIdx.x;
    int lane = tid & 31, wave = tid >> 5;
    int col  = lane & 15, hi = lane >> 4;
    int m0 = blockIdx.y * 64 + wave * 16;
    int n0 = blockIdx.x * 64;
    int arow = m0 + col; if (arow > M - 1) arow = M - 1;
    v8f acc[4];
    #pragma unroll
    for (int g = 0; g < 4; ++g) acc[g] = (v8f){0.f,0.f,0.f,0.f,0.f,0.f,0.f,0.f};
    const unsigned short* Abase = A + (size_t)arow * lda;
    const uint4* Arow = (const uint4*)Abase;
    for (int k0 = 0; k0 < K; k0 += 32){
        if (k0 + 128 < K)
            __builtin_prefetch((const void*)(Abase + k0 + 128), 0, 1);  // global_prefetch_b8
        int ka = (k0 >> 3) + hi * 2;      // A: K = hi*16+j
        v16bf af = make_frag(Arow[ka], Arow[ka + 1]);
        int kb = (k0 >> 3) + hi * 2;      // B: K = hi*16+j
        #pragma unroll
        for (int g = 0; g < 4; ++g){
            const uint4* Wrow = (const uint4*)(W + (size_t)(n0 + g * 16 + col) * ldw);
            v16bf bf = make_frag(Wrow[kb], Wrow[kb + 1]);
            acc[g] = wmma_bf16(af, bf, acc[g]);
        }
    }
    #pragma unroll
    for (int g = 0; g < 4; ++g){
        int n = n0 + g * 16 + col;
        #pragma unroll
        for (int r = 0; r < 8; ++r){
            int row = m0 + r + hi * 8;
            if (row < M){
                float v = acc[g][r];
                if (bias)  v += bias[n];
                if (bias2) v += bias2[n];
                if (epi == 1) v = 0.5f * v * (1.0f + erff(v * 0.70710678118654752f));
                size_t orow = row;
                if (tokens_per_b > 0){
                    int b = row / tokens_per_b;
                    orow = (size_t)b * L_ + tok_off + (row - b * tokens_per_b);
                }
                size_t idx = orow * (size_t)ldc + n;
                if (res) v += res[idx];
                if (Cb) Cb[idx] = f2bf(v); else Cf[idx] = v;
            }
        }
    }
}

// ---------- fused attention: one workgroup per (b, head), 4 waves ----------
__global__ void attn_kernel(const unsigned short* __restrict__ qt,
                            const unsigned short* __restrict__ kt,
                            const unsigned short* __restrict__ vt,
                            unsigned short* __restrict__ out){
    extern __shared__ __align__(16) char smem[];
    unsigned short* Ks  = (unsigned short*)smem;            // LP_ x 64 bf16
    unsigned short* Vts = Ks + LP_ * HD_;                   // 64 x LP_ bf16 (transposed V)
    float* Ssh = (float*)(smem + (size_t)2 * LP_ * HD_ * 2);// 4 waves x 16 x LP_ f32
    int tid  = threadIdx.x;
    int lane = tid & 31, wave = tid >> 5;
    int col  = lane & 15, hi = lane >> 4;
    size_t base = (size_t)blockIdx.x * L_ * HD_;
    int h = blockIdx.x % NH_;
    int b = blockIdx.x / NH_;

#if USE_TDM
    // stage K rows via Tensor Data Mover: 2D tile (341 rows x 64 bf16) -> LDS
    if (wave == 0){
        unsigned long long ga = (unsigned long long)(uintptr_t)(kt + base);
        unsigned lds_k = (unsigned)(uintptr_t)Ks;
        unsigned g0x = __builtin_amdgcn_readfirstlane(1u);                       // count=1
        unsigned g0y = __builtin_amdgcn_readfirstlane(lds_k);                    // lds_addr
        unsigned g0z = __builtin_amdgcn_readfirstlane((unsigned)(ga & 0xffffffffu));
        unsigned g0w = __builtin_amdgcn_readfirstlane(((unsigned)(ga >> 32) & 0x01ffffffu) | (2u << 30)); // type=2
        u32x4 g0 = { g0x, g0y, g0z, g0w };
        i32x8 g1 = { (int)(1u << 16),          // data_size = 2 bytes
                     (int)((unsigned)HD_ << 16),   // tensor_dim0[15:0] @ [31:16]
                     (int)((unsigned)L_  << 16),   // tensor_dim0 hi=0, tensor_dim1[15:0] @ [31:16]
                     (int)((unsigned)HD_ << 16),   // tensor_dim1 hi=0, tile_dim0 @ [31:16]
                     (int)(unsigned)L_,            // tile_dim1, tile_dim2=0
                     (int)(unsigned)HD_,           // tensor_dim0_stride low32
                     0, 0 };
        i32x4 z4 = {0, 0, 0, 0};
#if __clang_major__ >= 23
        i32x8 z8 = {0, 0, 0, 0, 0, 0, 0, 0};
        __builtin_amdgcn_tensor_load_to_lds(g0, g1, z4, z4, z8, 0);
#else
        __builtin_amdgcn_tensor_load_to_lds(g0, g1, z4, z4, 0);
#endif
        __builtin_amdgcn_s_wait_tensorcnt(0);
    }
#else
    for (int i = tid; i < L_ * 8; i += blockDim.x){
        int key = i >> 3, c = i & 7;
        ((uint4*)(Ks + key * HD_))[c] = ((const uint4*)(kt + base + (size_t)key * HD_))[c];
    }
#endif
    // zero K pad rows
    for (int i = tid; i < (LP_ - L_) * 8; i += blockDim.x){
        int key = L_ + (i >> 3), c = i & 7;
        ((uint4*)(Ks + key * HD_))[c] = make_uint4(0u, 0u, 0u, 0u);
    }
    // stage V transposed: Vts[dim*LP_ + key]
    for (int i = tid; i < L_ * 8; i += blockDim.x){
        int key = i >> 3, c = i & 7;
        uint4 d = ((const uint4*)(vt + base + (size_t)key * HD_))[c];
        const unsigned short* ds = (const unsigned short*)&d;
        #pragma unroll
        for (int j = 0; j < 8; ++j) Vts[(c * 8 + j) * LP_ + key] = ds[j];
    }
    for (int i = tid; i < (LP_ - L_) * HD_; i += blockDim.x){
        int key = L_ + i % (LP_ - L_);
        int dd  = i / (LP_ - L_);
        Vts[dd * LP_ + key] = 0;
    }
    __syncthreads();

    float* S = Ssh + wave * 16 * LP_;
    int nwaves = blockDim.x >> 5;

    for (int qtile = wave; qtile < LP_ / 16; qtile += nwaves){
        int q0 = qtile * 16;
        int qa = q0 + col; if (qa > L_ - 1) qa = L_ - 1;
        const uint4* Qrow = (const uint4*)(qt + base + (size_t)qa * HD_);
        v16bf qf0 = make_frag(Qrow[hi * 2],     Qrow[hi * 2 + 1]);     // HD chunk 0..31
        v16bf qf1 = make_frag(Qrow[4 + hi * 2], Qrow[5 + hi * 2]);     // HD chunk 32..63
        float rm[8];
        #pragma unroll
        for (int r = 0; r < 8; ++r) rm[r] = -3.0e38f;
        int qmax = q0 + 15; if (qmax > L_ - 1) qmax = L_ - 1;
        int ktn = (lvl_end(qmax) + 15) >> 4;

        for (int ktile = 0; ktile < LP_ / 16; ++ktile){
            v8f acc = (v8f){0.f,0.f,0.f,0.f,0.f,0.f,0.f,0.f};
            if (ktile < ktn){
                const uint4* Krow = (const uint4*)(Ks + (ktile * 16 + col) * HD_);
                acc = wmma_bf16(qf0, make_frag(Krow[hi * 2],     Krow[hi * 2 + 1]), acc);
                acc = wmma_bf16(qf1, make_frag(Krow[4 + hi * 2], Krow[5 + hi * 2]), acc);
            }
            int key = ktile * 16 + col;
            #pragma unroll
            for (int r = 0; r < 8; ++r){
                int q = q0 + hi * 8 + r;
                float s = acc[r] * 0.125f;                       // HD^-0.5
                if (ktile >= ktn || q > L_ - 1 || key >= lvl_end(q)) s = -1.0e30f;
                S[(hi * 8 + r) * LP_ + key] = s;
                rm[r] = fmaxf(rm[r], s);
            }
        }
        for (int r = 0; r < 8; ++r)
            for (int off = 1; off < 16; off <<= 1)
                rm[r] = fmaxf(rm[r], __shfl_xor(rm[r], off, 32));

        float rs[8];
        #pragma unroll
        for (int r = 0; r < 8; ++r) rs[r] = 0.f;
        for (int ktile = 0; ktile < LP_ / 16; ++ktile){
            #pragma unroll
            for (int r = 0; r < 8; ++r){
                int idx = (hi * 8 + r) * LP_ + ktile * 16 + col;
                float p = __expf(S[idx] - rm[r]);
                S[idx] = p;
                rs[r] += p;
            }
        }
        for (int r = 0; r < 8; ++r)
            for (int off = 1; off < 16; off <<= 1)
                rs[r] += __shfl_xor(rs[r], off, 32);

        v8f oacc[4];
        #pragma unroll
        for (int g = 0; g < 4; ++g) oacc[g] = (v8f){0.f,0.f,0.f,0.f,0.f,0.f,0.f,0.f};
        for (int kc2 = 0; kc2 < LP_ / 32; ++kc2){
            Frag16 pf;                                           // P in A-layout
            const float* Sr = S + col * LP_ + kc2 * 32 + hi * 16;
            #pragma unroll
            for (int j = 0; j < 8; ++j){ pf.s[j] = f2bf(Sr[j]); pf.s[8 + j] = f2bf(Sr[8 + j]); }
            #pragma unroll
            for (int g = 0; g < 4; ++g){
                const uint4* Vrow = (const uint4*)(Vts + (g * 16 + col) * LP_ + kc2 * 32 + hi * 16);
                oacc[g] = wmma_bf16(pf.v, make_frag(Vrow[0], Vrow[1]), oacc[g]);
            }
        }
        #pragma unroll
        for (int g = 0; g < 4; ++g){
            #pragma unroll
            for (int r = 0; r < 8; ++r){
                int q = q0 + hi * 8 + r;
                if (q < L_){
                    float v = oacc[g][r] / rs[r];
                    out[((size_t)b * L_ + q) * D_ + h * HD_ + g * 16 + col] = f2bf(v);
                }
            }
        }
    }
}

// ---------- host side ----------
static inline void conv(const void* in, void* out, int n, hipStream_t s){
    f2bf_kernel<<<(n + 255) / 256, 256, 0, s>>>((const float*)in, (unsigned short*)out, n);
}
static inline void gemm(const void* A, int lda, const void* W, int ldw,
                        float* Cf, unsigned short* Cb, int ldc,
                        const float* bias, const float* bias2, const float* res,
                        int M, int N, int K, int epi, int tpb, int tok_off, hipStream_t s){
    dim3 g(N / 64, (M + 63) / 64);
    gemm_bf16_kernel<<<g, 128, 0, s>>>((const unsigned short*)A, lda, (const unsigned short*)W, ldw,
                                       Cf, Cb, ldc, bias, bias2, res, M, N, K, epi, tpb, tok_off);
}

extern "C" void kernel_launch(void* const* d_in, const int* in_sizes, int n_in,
                              void* d_out, int out_size, void* d_ws, size_t ws_size,
                              hipStream_t stream){
    (void)in_sizes; (void)n_in; (void)out_size; (void)ws_size;
    auto F = [&](int i){ return (const float*)d_in[i]; };
    char* ws = (char*)d_ws;
    size_t off = 0;
    auto alloc = [&](size_t bytes)->char*{
        off = (off + 255) & ~(size_t)255;
        char* p = ws + off; off += bytes; return p;
    };

    // bf16 weights
    unsigned short* w1b = (unsigned short*)alloc((size_t)3072 * 512 * 2);
    unsigned short* w2b = (unsigned short*)alloc((size_t)768 * 3072 * 2);
    unsigned short* pwb[5];
    for (int i = 0; i < 5; ++i) pwb[i] = (unsigned short*)alloc((size_t)768 * 512 * 2);
    unsigned short *qkvw[6], *projw[6], *fc1w[6], *fc2w[6], *adaw[6];
    for (int i = 0; i < 6; ++i){
        qkvw[i] = (unsigned short*)alloc((size_t)2304 * 768 * 2);
        projw[i] = (unsigned short*)alloc((size_t)768 * 768 * 2);
        fc1w[i]  = (unsigned short*)alloc((size_t)3072 * 768 * 2);
        fc2w[i]  = (unsigned short*)alloc((size_t)768 * 3072 * 2);
        adaw[i]  = (unsigned short*)alloc((size_t)3072 * 768 * 2);
    }
    unsigned short* outw = (unsigned short*)alloc((size_t)512 * 768 * 2);

    // activations
    unsigned short* sv0b  = (unsigned short*)alloc((size_t)B_ * CIN_ * 2);
    unsigned short* shb   = (unsigned short*)alloc((size_t)B_ * 3072 * 2);
    float*          svf   = (float*)alloc((size_t)B_ * D_ * 4);
    unsigned short* svsb  = (unsigned short*)alloc((size_t)B_ * D_ * 2);
    unsigned short* poolb = (unsigned short*)alloc((size_t)B_ * L_ * CIN_ * 2);
    float*          x     = (float*)alloc((size_t)B_ * L_ * D_ * 4);
    float*          modb  = (float*)alloc((size_t)B_ * 3072 * 4);
    unsigned short* hb    = (unsigned short*)alloc((size_t)B_ * L_ * D_ * 2);
    unsigned short* qkvb  = (unsigned short*)alloc((size_t)B_ * L_ * 3 * D_ * 2);
    unsigned short* qtb   = (unsigned short*)alloc((size_t)B_ * L_ * D_ * 2);
    unsigned short* ktb   = (unsigned short*)alloc((size_t)B_ * L_ * D_ * 2);
    unsigned short* vtb   = (unsigned short*)alloc((size_t)B_ * L_ * D_ * 2);
    unsigned short* attb  = (unsigned short*)alloc((size_t)B_ * L_ * D_ * 2);
    unsigned short* midb  = (unsigned short*)alloc((size_t)B_ * L_ * 3072 * 2);
    unsigned short* finb  = (unsigned short*)alloc((size_t)B_ * 256 * D_ * 2);

    // weight conversions (fp32 params -> bf16)
    conv(F(2), w1b, 3072 * 512, stream);
    conv(F(4), w2b, 768 * 3072, stream);
    for (int i = 0; i < 5; ++i) conv(F(6 + 2 * i), pwb[i], 768 * 512, stream);
    for (int i = 0; i < 6; ++i){
        int pb = 17 + i * 13;
        conv(F(pb + 2),  qkvw[i], 2304 * 768, stream);
        conv(F(pb + 3),  projw[i], 768 * 768, stream);
        conv(F(pb + 7),  fc1w[i], 3072 * 768, stream);
        conv(F(pb + 9),  fc2w[i], 768 * 3072, stream);
        conv(F(pb + 11), adaw[i], 3072 * 768, stream);
    }
    conv(F(97), outw, 512 * 768, stream);

    // style MLP: sv = gelu(mean(style) W1^T + b1) W2^T + b2 ; silu(sv)
    style_pool_kernel<<<(B_ * CIN_ + 255) / 256, 256, 0, stream>>>(F(1), sv0b);
    gemm(sv0b, 512, w1b, 512, nullptr, shb, 3072, F(3), nullptr, nullptr, B_, 3072, 512, 1, 0, 0, stream);
    gemm(shb, 3072, w2b, 3072, svf, nullptr, 768, F(5), nullptr, nullptr, B_, 768, 3072, 0, 0, 0, stream);
    silu_kernel<<<(B_ * D_ + 255) / 256, 256, 0, stream>>>(svf, svsb, B_ * D_);

    // patch pooling + per-level embed (+lvl_embed) scattered into x
    static const int PNS[5]  = {1, 2, 4, 8, 16};
    static const int OFFS[5] = {0, 1, 5, 21, 85};
    for (int i = 0; i < 5; ++i){
        int T = PNS[i] * PNS[i];
        int total = B_ * T * CIN_;
        pool_kernel<<<(total + 255) / 256, 256, 0, stream>>>(F(0), poolb, PNS[i], B_ * OFFS[i]);
        gemm(poolb + (size_t)B_ * OFFS[i] * CIN_, 512, pwb[i], 512, x, nullptr, D_,
             F(7 + 2 * i), F(16) + i * D_, nullptr, B_ * T, D_, 512, 0, T, OFFS[i], stream);
    }

    // transformer blocks
    for (int i = 0; i < 6; ++i){
        int pb = 17 + i * 13;
        gemm(svsb, 768, adaw[i], 768, modb, nullptr, 3072, F(pb + 12), nullptr, nullptr,
             B_, 3072, 768, 0, 0, 0, stream);
        ln_kernel<<<B_ * L_, 256, 0, stream>>>(x, F(pb + 0), F(pb + 1), modb, 0, 768, hb, L_, 0, 1e-6f);
        gemm(hb, 768, qkvw[i], 768, nullptr, qkvb, 2304, nullptr, nullptr, nullptr,
             B_ * L_, 2304, 768, 0, 0, 0, stream);
        rope_kernel<<<B_ * L_, 384, 0, stream>>>(qkvb, qtb, ktb, vtb);
        attn_kernel<<<B_ * NH_, 128, 180224, stream>>>(qtb, ktb, vtb, attb);
        gemm(attb, 768, projw[i], 768, x, nullptr, D_, F(pb + 4), nullptr, x,
             B_ * L_, D_, 768, 0, 0, 0, stream);
        ln_kernel<<<B_ * L_, 256, 0, stream>>>(x, F(pb + 5), F(pb + 6), modb, 1536, 2304, hb, L_, 0, 1e-6f);
        gemm(hb, 768, fc1w[i], 768, nullptr, midb, 3072, F(pb + 8), nullptr, nullptr,
             B_ * L_, 3072, 768, 1, 0, 0, stream);
        gemm(midb, 3072, fc2w[i], 3072, x, nullptr, D_, F(pb + 10), nullptr, x,
             B_ * L_, D_, 3072, 0, 0, 0, stream);
    }

    // final LN over last 256 tokens + output projection
    ln_kernel<<<B_ * 256, 256, 0, stream>>>(x, F(95), F(96), nullptr, 0, 0, finb, 256, 85, 1e-5f);
    gemm(finb, 768, outw, 768, (float*)d_out, nullptr, OUTD_, F(98), nullptr, nullptr,
         B_ * 256, OUTD_, 768, 0, 0, 0, stream);
}
